// DCNConvModule_4200478015845
// MI455X (gfx1250) — matrix operand order repaired and checked by reading, not compiled, not run on hardware
//
#include <hip/hip_runtime.h>
#include <hip/hip_bf16.h>

typedef __attribute__((ext_vector_type(16))) __bf16 v16bf;
typedef __attribute__((ext_vector_type(8)))  float  v8f;

#if __has_builtin(__builtin_amdgcn_global_load_async_to_lds_b128) && \
    __has_builtin(__builtin_amdgcn_global_load_async_to_lds_b64)  && \
    __has_builtin(__builtin_amdgcn_s_wait_asynccnt)
#define HAVE_ASYNC_LDS 1
#else
#define HAVE_ASYNC_LDS 0
#endif

// exact parameter types from hipcc diagnostics: GCC-style int vectors,
// AS(1) (printed as __device__) source, AS(3) LDS destination
typedef int a2i __attribute__((vector_size(8)));
typedef int a4i __attribute__((vector_size(16)));
#define GP64(p)  ((__attribute__((address_space(1))) a2i*)(p))
#define LP64(p)  ((__attribute__((address_space(3))) a2i*)(p))
#define GP128(p) ((__attribute__((address_space(1))) a4i*)(p))
#define LP128(p) ((__attribute__((address_space(3))) a4i*)(p))

#define B_    4
#define C_    256
#define H_    64
#define W_    64
#define HW    4096
#define CO_   256
#define KK_   9
#define KDIM  2304     // C_ * KK_   (K ordering: k' = kk*256 + c, tap-major)
#define KSTEP 32
#define NTILE 64       // main GEMM N tile
#define NALL  16384    // B_*HW

struct SampleInfo { int idx[4]; float wgt[4]; };   // 32 bytes

// workspace layout (bytes)
#define WS_SINFO   0          //  4,718,592  SampleInfo[4*9*4096]
#define WS_WBF     4718592    //  1,179,648  bf16 [256][2304]  (tap-major K)
#define WS_WOFFBF  5898240    //    147,456  bf16 [32][2304]   (tap-major K, zero pad)
#define WS_OUTOFF  6045696    //  1,769,472  f32  [27][16384]
#define WS_PART    7815168    //      2,048  f32  [256][2]
#define WS_STATS   7817216    //         32  f32  [4][2]

// ---------------------------------------------------------------------------
// Kernel A: main conv weights fp32 -> bf16, permuted [M=256][k'=kk*256+c]
// ---------------------------------------------------------------------------
__global__ void dcn_cvt_weights(const float* __restrict__ w,
                                __bf16* __restrict__ wbf)
{
    int i = blockIdx.x * blockDim.x + threadIdx.x;
    if (i >= CO_ * KDIM) return;
    int o  = i / KDIM;
    int kp = i - o * KDIM;
    int kk = kp >> 8;          // tap
    int c  = kp & 255;         // channel
    wbf[i] = (__bf16)w[(size_t)o * KDIM + c * KK_ + kk];
}

// ---------------------------------------------------------------------------
// Kernel B: offset conv weights fp32 -> bf16, permuted padded [32][2304]
// ---------------------------------------------------------------------------
__global__ void dcn_cvt_woff(const float* __restrict__ w_off,
                             __bf16* __restrict__ woffbf)
{
    int i = blockIdx.x * blockDim.x + threadIdx.x;
    if (i >= 32 * KDIM) return;
    int row = i / KDIM;
    int kp  = i - row * KDIM;
    int kk  = kp >> 8;
    int c   = kp & 255;
    woffbf[i] = (row < 27) ? (__bf16)w_off[(size_t)row * KDIM + c * KK_ + kk]
                           : (__bf16)0.f;
}

// ---------------------------------------------------------------------------
// Kernel C: offset conv as WMMA GEMM  (M=32, K=2304, N=16384)
//   block: 256 threads, tile M=32 x N=256; im2col built on the fly.
//   K chunk of 32 lies inside one tap block -> tap uniform per step.
// ---------------------------------------------------------------------------
__global__ void __launch_bounds__(256, 1)
dcn_off_gemm(const float* __restrict__ x,
             const __bf16* __restrict__ woffbf,
             const float* __restrict__ b_off,
             float* __restrict__ out_off)        // [27][16384]
{
    __shared__ __bf16 At[32][32];     //  2 KB [m][k]
    __shared__ __bf16 Bt[256][32];    // 16 KB [n][k]

    const int bid  = blockIdx.x;      // 64 blocks
    const int b    = bid >> 4;
    const int p0   = (bid & 15) * 256;
    const int t    = threadIdx.x;
    const int lane = t & 31;
    const int wave = t >> 5;

    const float* xb = x + (size_t)b * C_ * HW;
    const int p = p0 + t, h = p >> 6, w = p & 63;

    v8f acc[2][2] = {};

    for (int kb = 0; kb < KDIM; kb += KSTEP) {
        // A tile: each thread copies 4 bf16 (8 B)
        {
            int idx = t * 4;
            int row = idx >> 5, col = idx & 31;
            const __bf16* src = woffbf + (size_t)row * KDIM + kb + col;
#if HAVE_ASYNC_LDS
            __builtin_amdgcn_global_load_async_to_lds_b64(
                GP64(src), LP64(&At[row][col]), 0, 0);
#else
            *(uint2*)&At[row][col] = *(const uint2*)src;
#endif
        }
        // B tile: tap uniform for this chunk; 32 channel loads, imm offsets
        {
            int tap = kb >> 8;
            int c0  = kb & 255;
            int yy = h - 1 + tap / 3;
            int xx = w - 1 + tap % 3;
            bool tv = (yy >= 0) && (yy < H_) && (xx >= 0) && (xx < W_);
            if (tv) {
                const float* xc = xb + (size_t)c0 * HW + yy * W_ + xx;
#pragma unroll
                for (int kq = 0; kq < 32; ++kq)
                    Bt[t][kq] = (__bf16)xc[(size_t)kq * HW];
            } else {
#pragma unroll
                for (int kq = 0; kq < 32; ++kq)
                    Bt[t][kq] = (__bf16)0.f;
            }
        }
#if HAVE_ASYNC_LDS
        __builtin_amdgcn_s_wait_asynccnt(0);
#endif
        __syncthreads();

        const int rlo   = lane & 15;
        const int akoff = (lane & 16) ? 8 : 0;
        const int bkoff = (lane & 16) ? 8 : 0;
        union Frag { uint4 q[2]; v16bf v; };
        Frag af[2], bf[2];
#pragma unroll
        for (int mi = 0; mi < 2; ++mi) {
            const __bf16* ap = &At[mi * 16 + rlo][akoff];
            af[mi].q[0] = *(const uint4*)ap;
            af[mi].q[1] = *(const uint4*)(ap + 16);
        }
#pragma unroll
        for (int ni = 0; ni < 2; ++ni) {
            const __bf16* bp = &Bt[(wave * 2 + ni) * 16 + rlo][bkoff];
            bf[ni].q[0] = *(const uint4*)bp;
            bf[ni].q[1] = *(const uint4*)(bp + 16);
        }
#pragma unroll
        for (int mi = 0; mi < 2; ++mi)
#pragma unroll
            for (int ni = 0; ni < 2; ++ni)
                acc[mi][ni] = __builtin_amdgcn_wmma_f32_16x16x32_bf16(
                    false, af[mi].v, false, bf[ni].v,
                    (short)0, acc[mi][ni], false, false);
        __syncthreads();
    }

    // epilogue: add bias, store rows 0..26 to out_off[row][b*HW + p]
    const int col  = lane & 15;
    const int rsel = (lane & 16) ? 8 : 0;
#pragma unroll
    for (int mi = 0; mi < 2; ++mi) {
#pragma unroll
        for (int ni = 0; ni < 2; ++ni) {
#pragma unroll
            for (int r = 0; r < 8; ++r) {
                int row = mi * 16 + rsel + r;
                if (row < 27) {
                    int nloc = (wave * 2 + ni) * 16 + col;
                    out_off[(size_t)row * NALL + (size_t)b * HW + p0 + nloc] =
                        acc[mi][ni][r] + b_off[row];
                }
            }
        }
    }
}

// ---------------------------------------------------------------------------
// Kernel D: (dy,dx,mask) -> packed bilinear SampleInfo
// ---------------------------------------------------------------------------
__global__ void dcn_prep_coeff(const float* __restrict__ out_off,
                               SampleInfo* __restrict__ sinfo)
{
    int tid = blockIdx.x * blockDim.x + threadIdx.x;
    if (tid >= B_ * KK_ * HW) return;
    int p  = tid & (HW - 1);
    int kk = (tid >> 12) % KK_;
    int b  = tid / (KK_ * HW);
    int h = p >> 6, w = p & 63;

    int n = b * HW + p;
    float ady = out_off[(size_t)(2 * kk    ) * NALL + n];
    float adx = out_off[(size_t)(2 * kk + 1) * NALL + n];
    float am  = out_off[(size_t)(18 + kk   ) * NALL + n];

    float m = 1.0f / (1.0f + __expf(-am));
    int ki = kk / 3, kj = kk % 3;
    float py = (float)(h - 1 + ki) + ady;
    float px = (float)(w - 1 + kj) + adx;
    float y0f = floorf(py), x0f = floorf(px);
    float ly = py - y0f, lx = px - x0f;
    int y0 = (int)y0f, x0 = (int)x0f;

    float cw[4] = { (1.f - ly) * (1.f - lx), (1.f - ly) * lx,
                    ly * (1.f - lx),         ly * lx };
    int cy[4] = { y0, y0, y0 + 1, y0 + 1 };
    int cx[4] = { x0, x0 + 1, x0, x0 + 1 };

    SampleInfo s;
#pragma unroll
    for (int j = 0; j < 4; ++j) {
        bool v = (cy[j] >= 0) && (cy[j] < H_) && (cx[j] >= 0) && (cx[j] < W_);
        int yc = min(max(cy[j], 0), H_ - 1);
        int xc = min(max(cx[j], 0), W_ - 1);
        s.idx[j] = yc * W_ + xc;
        s.wgt[j] = v ? cw[j] * m : 0.0f;   // fold mask into bilinear weights
    }
    sinfo[tid] = s;
}

// ---------------------------------------------------------------------------
// Kernel E: fused deformable-gather + WMMA GEMM + bias + GN partial sums
//   block: 256 threads (8 waves), tile M=256 x N=64, K-loop step 32.
//   Tap-major K: one SampleInfo load per thread per K-step.
// ---------------------------------------------------------------------------
__global__ void __launch_bounds__(256, 1)
dcn_gemm_wmma(const float* __restrict__ x,
              const __bf16* __restrict__ wbf,
              const SampleInfo* __restrict__ sinfo,
              const float* __restrict__ bias,
              float* __restrict__ out,        // [B][CO][HW]
              float* __restrict__ partials)   // [256][2]
{
    __shared__ __bf16 Atile[256][32];   // 16 KB  [M][K]
    __shared__ __bf16 Btile[64][32];    //  4 KB  [N][K]
    __shared__ float  red[256];

    const int bid  = blockIdx.x;            // 0..255
    const int b    = bid >> 6;              // batch
    const int p0   = (bid & 63) * NTILE;    // pixel start within batch
    const int t    = threadIdx.x;
    const int lane = t & 31;
    const int wave = t >> 5;

    const float* xb = x + (size_t)b * C_ * HW;
    const SampleInfo* sb = sinfo + (size_t)b * KK_ * HW;

    v8f acc[2][4] = {};

    const int bn  = t & 63;          // B-build: pixel column 0..63
    const int bk8 = (t >> 6) * 8;    // B-build: 8 consecutive k

    for (int kb = 0; kb < KDIM; kb += KSTEP) {
        // ---- stage A tile: thread t stages its weight row's 32 bf16 (64 B)
        {
            const __bf16* src = wbf + (size_t)t * KDIM + kb;
#if HAVE_ASYNC_LDS
            __builtin_amdgcn_global_load_async_to_lds_b128(
                GP128(src), LP128(&Atile[t][0]), 0, 0);
            __builtin_amdgcn_global_load_async_to_lds_b128(
                GP128(src), LP128(&Atile[t][0]), 16, 0);
            __builtin_amdgcn_global_load_async_to_lds_b128(
                GP128(src), LP128(&Atile[t][0]), 32, 0);
            __builtin_amdgcn_global_load_async_to_lds_b128(
                GP128(src), LP128(&Atile[t][0]), 48, 0);
#else
            uint4* dst = (uint4*)&Atile[t][0];
            const uint4* s4 = (const uint4*)src;
            dst[0] = s4[0]; dst[1] = s4[1]; dst[2] = s4[2]; dst[3] = s4[3];
#endif
            if (kb + KSTEP < KDIM)
                __builtin_prefetch((const void*)(src + KSTEP), 0, 1); // global_prefetch_b8
        }
        // ---- stage B tile: kk uniform per chunk; 1 SampleInfo + 32 gathers
        {
            int kk = kb >> 8;              // tap index (uniform)
            int c0 = (kb & 255) + bk8;     // first channel for this thread
            SampleInfo s = sb[kk * HW + p0 + bn];
            const float* xp0 = xb + (size_t)c0 * HW;
            const float* g0 = xp0 + s.idx[0];
            const float* g1 = xp0 + s.idx[1];
            const float* g2 = xp0 + s.idx[2];
            const float* g3 = xp0 + s.idx[3];
#pragma unroll
            for (int i = 0; i < 8; ++i) {
                float v = s.wgt[0] * g0[(size_t)i * HW]
                        + s.wgt[1] * g1[(size_t)i * HW]
                        + s.wgt[2] * g2[(size_t)i * HW]
                        + s.wgt[3] * g3[(size_t)i * HW];
                Btile[bn][bk8 + i] = (__bf16)v;
            }
        }
#if HAVE_ASYNC_LDS
        __builtin_amdgcn_s_wait_asynccnt(0);
#endif
        __syncthreads();

        // ---- WMMA: wave handles msub {2w, 2w+1} x nsub {0..3}
        const int m0    = wave * 2;
        const int rlo   = lane & 15;
        const int akoff = (lane & 16) ? 8 : 0;   // A: lanes 16-31 hold K=8..15,24..31
        const int bkoff = (lane & 16) ? 8 : 0;   // B: lanes 16-31 hold K=8..15,24..31
        union Frag { uint4 q[2]; v16bf v; };
        Frag afr[2], bfr[4];
#pragma unroll
        for (int mi = 0; mi < 2; ++mi) {
            const __bf16* ap = &Atile[(m0 + mi) * 16 + rlo][akoff];
            afr[mi].q[0] = *(const uint4*)ap;
            afr[mi].q[1] = *(const uint4*)(ap + 16);
        }
#pragma unroll
        for (int ni = 0; ni < 4; ++ni) {
            const __bf16* bp = &Btile[ni * 16 + rlo][bkoff];
            bfr[ni].q[0] = *(const uint4*)bp;
            bfr[ni].q[1] = *(const uint4*)(bp + 16);
        }
#pragma unroll
        for (int mi = 0; mi < 2; ++mi)
#pragma unroll
            for (int ni = 0; ni < 4; ++ni)
                acc[mi][ni] = __builtin_amdgcn_wmma_f32_16x16x32_bf16(
                    false, afr[mi].v, false, bfr[ni].v,
                    (short)0, acc[mi][ni], false, false);
        __syncthreads();
    }

    // ---- epilogue: bias, store, per-block GN partial sums
    float lsum = 0.f, lsq = 0.f;
    const int m0   = wave * 2;
    const int col  = lane & 15;
    const int rsel = (lane & 16) ? 8 : 0;   // C/D layout: lanes 16-31 hold M=8..15
#pragma unroll
    for (int mi = 0; mi < 2; ++mi) {
#pragma unroll
        for (int ni = 0; ni < 4; ++ni) {
#pragma unroll
            for (int r = 0; r < 8; ++r) {
                int row = (m0 + mi) * 16 + rsel + r;
                float v = acc[mi][ni][r] + bias[row];
                out[((size_t)b * CO_ + row) * HW + p0 + ni * 16 + col] = v;
                lsum += v;
                lsq = fmaf(v, v, lsq);
            }
        }
    }
    red[t] = lsum; __syncthreads();
    for (int s = 128; s > 0; s >>= 1) { if (t < s) red[t] += red[t + s]; __syncthreads(); }
    float bsum = red[0]; __syncthreads();
    red[t] = lsq; __syncthreads();
    for (int s = 128; s > 0; s >>= 1) { if (t < s) red[t] += red[t + s]; __syncthreads(); }
    if (t == 0) { partials[bid * 2 + 0] = bsum; partials[bid * 2 + 1] = red[0]; }
}

// ---------------------------------------------------------------------------
// Kernel F: deterministic reduction of per-block partials -> mean, rstd
// ---------------------------------------------------------------------------
__global__ void dcn_gn_stats(const float* __restrict__ partials,
                             float* __restrict__ stats)
{
    __shared__ float s0[64], s1[64];
    int b = blockIdx.x, t = threadIdx.x;    // 64 threads
    s0[t] = partials[(b * 64 + t) * 2 + 0];
    s1[t] = partials[(b * 64 + t) * 2 + 1];
    __syncthreads();
    for (int s = 32; s > 0; s >>= 1) {
        if (t < s) { s0[t] += s0[t + s]; s1[t] += s1[t + s]; }
        __syncthreads();
    }
    if (t == 0) {
        const float invN = 1.0f / (float)(CO_ * HW);
        float mean = s0[0] * invN;
        float var  = s1[0] * invN - mean * mean;
        stats[b * 2 + 0] = mean;
        stats[b * 2 + 1] = rsqrtf(var + 1e-5f);
    }
}

// ---------------------------------------------------------------------------
// Kernel G: in-place GroupNorm affine + ReLU (float4 traffic)
// ---------------------------------------------------------------------------
__global__ void dcn_gn_apply(float* __restrict__ out,
                             const float* __restrict__ stats,
                             const float* __restrict__ gamma,
                             const float* __restrict__ beta)
{
    int i4 = (blockIdx.x * blockDim.x + threadIdx.x) * 4;
    int b = i4 >> 20;            // CO_*HW = 2^20
    int o = (i4 >> 12) & 255;    // HW = 2^12
    float mean = stats[b * 2 + 0], rstd = stats[b * 2 + 1];
    float g  = gamma[o] * rstd;
    float be = beta[o] - mean * g;      // (v-mean)*rstd*gamma+beta = v*g + be
    float4 v = *(float4*)(out + i4);
    v.x = fmaxf(fmaf(v.x, g, be), 0.f);
    v.y = fmaxf(fmaf(v.y, g, be), 0.f);
    v.z = fmaxf(fmaf(v.z, g, be), 0.f);
    v.w = fmaxf(fmaf(v.w, g, be), 0.f);
    *(float4*)(out + i4) = v;
}

// ---------------------------------------------------------------------------
extern "C" void kernel_launch(void* const* d_in, const int* in_sizes, int n_in,
                              void* d_out, int out_size, void* d_ws, size_t ws_size,
                              hipStream_t stream)
{
    const float* x     = (const float*)d_in[0];
    const float* w_off = (const float*)d_in[1];
    const float* b_off = (const float*)d_in[2];
    const float* w     = (const float*)d_in[3];
    const float* bias  = (const float*)d_in[4];
    const float* gamma = (const float*)d_in[5];
    const float* beta  = (const float*)d_in[6];
    float* out = (float*)d_out;

    char* ws = (char*)d_ws;
    SampleInfo* sinfo  = (SampleInfo*)(ws + WS_SINFO);
    __bf16*     wbf    = (__bf16*)(ws + WS_WBF);
    __bf16*     woffbf = (__bf16*)(ws + WS_WOFFBF);
    float*      outoff = (float*)(ws + WS_OUTOFF);
    float*      parts  = (float*)(ws + WS_PART);
    float*      stats  = (float*)(ws + WS_STATS);

    dcn_cvt_weights<<<2304, 256, 0, stream>>>(w, wbf);
    dcn_cvt_woff   <<<288, 256, 0, stream>>>(w_off, woffbf);
    dcn_off_gemm   <<<64, 256, 0, stream>>>(x, woffbf, b_off, outoff);
    dcn_prep_coeff <<<576, 256, 0, stream>>>(outoff, sinfo);
    dcn_gemm_wmma  <<<256, 256, 0, stream>>>(x, wbf, sinfo, bias, out, parts);
    dcn_gn_stats   <<<4, 64, 0, stream>>>(parts, stats);
    dcn_gn_apply   <<<4096, 256, 0, stream>>>(out, stats, gamma, beta);
}
